// GraphAttentionLayer_65214783422561
// MI455X (gfx1250) — matrix operand order repaired and checked by reading, not compiled, not run on hardware
//
#include <hip/hip_runtime.h>

// -------- GAT layer: B=4, N=4096, F_IN=F_OUT=128, leaky_relu slope 0.2 -----
// kernel1: Wh = h@W via v_wmma_f32_16x16x32_bf16 (f32 accum), computes
//          s1 = Wh.a1, s2 = Wh.a2, stores Wh^T (f-major) bf16 to workspace.
// kernel2: flash-attention style fused softmax(leakyrelu(s1_i+s2_j)) @ Wh.
//          The 128x128 bf16 Wh^T tile per j-block is DMA'd into LDS by the
//          Tensor Data Mover (tensor_load_to_lds + s_wait_tensorcnt), then
//          P@Wh tiles run on WMMA with online-softmax rescaling.

#define NB 4
#define NN 4096
#define FF 128
#define ALPHA 0.2f
#define LOG2E 1.4426950408889634f

typedef __attribute__((ext_vector_type(16))) __bf16 v16bf;
typedef __attribute__((ext_vector_type(8)))  float  v8f;
typedef __attribute__((ext_vector_type(4)))  unsigned int v4u;
typedef __attribute__((ext_vector_type(8)))  unsigned int v8u;

// native f32 -> bf16 (RNE) — lowers to v_cvt_pk_bf16_f32-class ops on gfx1250
__device__ __forceinline__ __bf16 f2bf(float f) { return (__bf16)f; }
__device__ __forceinline__ unsigned short f2bf_bits(float f) {
  __bf16 b = (__bf16)f;
  unsigned short u;
  __builtin_memcpy(&u, &b, 2);
  return u;
}

// A-frag (16x32 bf16) per-lane K mapping (ISA 7.12.2):
// lane = m + 16*h ; slots e=0..7  -> K = 8h + e
//                   slots e=8..15 -> K = 16 + 8h + (e-8)
// B-frag (32x16 bf16): lane = n + 16*h ; slot e -> same K as A-frag slot e
// C/D (16x16 f32): VGPR r: lanes 0-15 -> M=r, lanes 16-31 -> M=r+8, N=lane&15

// ------------------------------ kernel 1 -----------------------------------
__global__ __launch_bounds__(256) void gat_wh_kernel(
    const float* __restrict__ h, const float* __restrict__ W,
    const float* __restrict__ a,
    unsigned short* __restrict__ WhT,   // [b][f][jloc] bf16, b*128*4096
    float* __restrict__ s1w, float* __restrict__ s2w) {
  __shared__ __align__(16) unsigned short Wt[FF * FF];  // Wt[o*128 + f]
  const int tid = threadIdx.x;

  // stage W transposed (col-major) as bf16 into LDS
  for (int i = tid; i < FF * FF; i += 256) {
    int f = i >> 7, o = i & 127;
    Wt[o * FF + f] = f2bf_bits(W[i]);
  }
  __syncthreads();

  const int wave = tid >> 5, lane = tid & 31;
  const int n = lane & 15, hh = lane >> 4;
  const int base_m = blockIdx.x * 128 + wave * 16;
  const int rowA = base_m + n;  // A-matrix row handled by this lane

  v8f acc[8] = {};

#pragma unroll
  for (int kk = 0; kk < 4; ++kk) {
    const int k0 = kk * 32;
    // two contiguous 8-float runs per lane (per A-frag K mapping)
    const float4* p0 = (const float4*)(h + (size_t)rowA * FF + k0 + 8 * hh);
    const float4* p1 = (const float4*)(h + (size_t)rowA * FF + k0 + 16 + 8 * hh);
    float4 f0 = p0[0], f1 = p0[1], f2 = p1[0], f3 = p1[1];
    float av[16] = {f0.x, f0.y, f0.z, f0.w, f1.x, f1.y, f1.z, f1.w,
                    f2.x, f2.y, f2.z, f2.w, f3.x, f3.y, f3.z, f3.w};
    v16bf afrag;
#pragma unroll
    for (int e = 0; e < 16; ++e) afrag[e] = f2bf(av[e]);

#pragma unroll
    for (int t = 0; t < 8; ++t) {
      v16bf bfrag;
      __builtin_memcpy(&bfrag, &Wt[(16 * t + n) * FF + k0 + 8 * hh], 16);
      __builtin_memcpy((char*)&bfrag + 16, &Wt[(16 * t + n) * FF + k0 + 16 + 8 * hh], 16);
      acc[t] = __builtin_amdgcn_wmma_f32_16x16x32_bf16(
          false, afrag, false, bfrag, (short)0, acc[t], false, false);
    }
  }

  // epilogue: s1/s2 + transposed bf16 Wh store
  float a1v[8], a2v[8];
#pragma unroll
  for (int t = 0; t < 8; ++t) {
    a1v[t] = a[16 * t + n];
    a2v[t] = a[FF + 16 * t + n];
  }
  const int b = base_m >> 12;
#pragma unroll
  for (int r = 0; r < 8; ++r) {
    const int grow = base_m + r + 8 * hh;   // global row for this acc VGPR
    const int jloc = grow & 4095;
    float s1p = 0.f, s2p = 0.f;
#pragma unroll
    for (int t = 0; t < 8; ++t) {
      float v = acc[t][r];
      s1p += v * a1v[t];
      s2p += v * a2v[t];
      WhT[((size_t)b * FF + 16 * t + n) * NN + jloc] = f2bf_bits(v);
    }
#pragma unroll
    for (int off = 1; off < 16; off <<= 1) {
      s1p += __shfl_xor(s1p, off);
      s2p += __shfl_xor(s2p, off);
    }
    if (n == 0) { s1w[grow] = s1p; s2w[grow] = s2p; }
  }
}

// ------------------------------ kernel 2 -----------------------------------
__global__ __launch_bounds__(256) void gat_attn_kernel(
    const unsigned short* __restrict__ WhT, const float* __restrict__ s1w,
    const float* __restrict__ s2w, float* __restrict__ out) {
  __shared__ __align__(16) unsigned short Vt[FF * 128];  // Vt[f*128 + jloc]
  __shared__ float s2s[128];

  const int tid = threadIdx.x, wave = tid >> 5, lane = tid & 31;
  const int n = lane & 15, hh = lane >> 4;
  const int base_m = blockIdx.x * 128 + wave * 16;
  const int b = (blockIdx.x * 128) >> 12;
  const unsigned short* Vsrc = WhT + (size_t)b * FF * NN;
  const int jbatch0 = b << 12;

  const float s1_lane = s1w[base_m + n];
  float m_run = -__builtin_inff(), l_run = 0.f;
  v8f acc[8] = {};

  // ---- TDM descriptor group 1 (constant across j-blocks) -------------------
  // 2D tile: tile_dim0=128 (j, contiguous), tile_dim1=128 (f rows),
  // tensor_dim0=4096, tensor_dim1=128, tensor_dim0_stride=4096, data_size=2B.
  v8u g1;
  g1[0] = 1u << 16;          // wg_mask=0 | data_size=1 (2 bytes)
  g1[1] = 4096u << 16;       // tensor_dim0[15:0] in bits[31:16]
  g1[2] = 128u << 16;        // tensor_dim0 hi=0 | tensor_dim1[15:0]
  g1[3] = 128u << 16;        // tensor_dim1 hi=0 | tile_dim0=128
  g1[4] = 128u;              // tile_dim1=128 | tile_dim2=0
  g1[5] = 4096u;             // tensor_dim0_stride[31:0]
  g1[6] = 0u;                // stride hi | tensor_dim1_stride lo
  g1[7] = 0u;                // tensor_dim1_stride hi
  const unsigned ldsVt = (unsigned)(size_t)(void*)&Vt[0];

  for (int jb = 0; jb < 32; ++jb) {
    const int j0 = jb * 128;
    __syncthreads();  // previous block fully consumed

    if (wave == 0) {  // one TDM descriptor per workgroup per j-block
      unsigned long long ga =
          (unsigned long long)(size_t)(const void*)Vsrc +
          (unsigned long long)j0 * 2ull;
      v4u g0;
      g0[0] = 1u;                                   // count=1 (valid user D#)
      g0[1] = ldsVt;                                // lds_addr (bytes)
      g0[2] = (unsigned)ga;                         // global_addr[31:0]
      g0[3] = (unsigned)((ga >> 32) & 0x01FFFFFFu)  // global_addr[56:32]
              | 0x80000000u;                        // type=2 ("image")
      asm volatile("tensor_load_to_lds %0, %1" :: "s"(g0), "s"(g1) : "memory");
      __builtin_amdgcn_s_wait_tensorcnt(0);
    }
    if (tid < 128) s2s[tid] = s2w[jbatch0 + j0 + tid];
    __syncthreads();

    if (jb + 1 < 32) {  // warm L2 for next block (global_prefetch_b8)
      const int f = tid >> 1, half = (tid & 1) * 64;
      __builtin_prefetch(Vsrc + (size_t)f * NN + j0 + 128 + half, 0, 0);
    }

    // e-tile in A-frag layout; find block max
    float ev[64];
    float m_blk = -__builtin_inff();
#pragma unroll
    for (int kk = 0; kk < 4; ++kk) {
      const int kb = kk * 32;
      const float4* q0 = (const float4*)&s2s[kb + 8 * hh];
      const float4* q1 = (const float4*)&s2s[kb + 16 + 8 * hh];
      float4 v0 = q0[0], v1 = q0[1], v2 = q1[0], v3 = q1[1];
      float sv[16] = {v0.x, v0.y, v0.z, v0.w, v1.x, v1.y, v1.z, v1.w,
                      v2.x, v2.y, v2.z, v2.w, v3.x, v3.y, v3.z, v3.w};
#pragma unroll
      for (int e = 0; e < 16; ++e) {
        float x = s1_lane + sv[e];
        x = (x >= 0.f) ? x : ALPHA * x;   // leaky_relu
        ev[kk * 16 + e] = x;
        m_blk = fmaxf(m_blk, x);
      }
    }
    m_blk = fmaxf(m_blk, __shfl_xor(m_blk, 16));  // combine row halves
    const float m_new = fmaxf(m_run, m_blk);
    const float scale = exp2f((m_run - m_new) * LOG2E);
    l_run *= scale;
#pragma unroll
    for (int r = 0; r < 8; ++r) {
      const float sc = __shfl(scale, r + 8 * hh);  // stat for row r+8*hh
#pragma unroll
      for (int t = 0; t < 8; ++t) acc[t][r] *= sc;
    }

#pragma unroll
    for (int kk = 0; kk < 4; ++kk) {
      v16bf pfrag;
      float psum = 0.f;
#pragma unroll
      for (int e = 0; e < 16; ++e) {
        float p = exp2f((ev[kk * 16 + e] - m_new) * LOG2E);
        psum += p;
        pfrag[e] = f2bf(p);
      }
      l_run += psum;
#pragma unroll
      for (int t = 0; t < 8; ++t) {
        v16bf bfrag;
        __builtin_memcpy(&bfrag, &Vt[(16 * t + n) * 128 + kk * 32 + 8 * hh], 16);
        __builtin_memcpy((char*)&bfrag + 16, &Vt[(16 * t + n) * 128 + kk * 32 + 16 + 8 * hh], 16);
        acc[t] = __builtin_amdgcn_wmma_f32_16x16x32_bf16(
            false, pfrag, false, bfrag, (short)0, acc[t], false, false);
      }
    }
    m_run = m_new;
  }

  // finalize: divide by row sum, ELU, store
  const float l_tot = l_run + __shfl_xor(l_run, 16);
  const float inv = 1.0f / l_tot;
#pragma unroll
  for (int r = 0; r < 8; ++r) {
    const float invr = __shfl(inv, r + 8 * hh);
    const int grow = base_m + r + 8 * hh;
#pragma unroll
    for (int t = 0; t < 8; ++t) {
      float v = acc[t][r] * invr;
      v = (v > 0.f) ? v : (exp2f(v * LOG2E) - 1.0f);  // elu
      out[(size_t)grow * FF + 16 * t + n] = v;
    }
  }
}

// ------------------------------ launcher -----------------------------------
extern "C" void kernel_launch(void* const* d_in, const int* in_sizes, int n_in,
                              void* d_out, int out_size, void* d_ws,
                              size_t ws_size, hipStream_t stream) {
  const float* h = (const float*)d_in[0];   // [4,4096,128] f32
  const float* W = (const float*)d_in[1];   // [128,128] f32
  const float* a = (const float*)d_in[2];   // [256] f32
  float* out = (float*)d_out;               // [4,4096,128] f32

  // workspace: Wh^T bf16 (4 MB) + s1 + s2 (64 KB each)
  unsigned short* WhT = (unsigned short*)d_ws;
  float* s1w = (float*)((char*)d_ws + (size_t)4 * 1024 * 1024);
  float* s2w = s1w + NB * NN;

  gat_wh_kernel<<<128, 256, 0, stream>>>(h, W, a, WhT, s1w, s2w);
  gat_attn_kernel<<<128, 256, 0, stream>>>(WhT, s1w, s2w, out);
}
